// ContinuousActor_30691836297669
// MI455X (gfx1250) — matrix operand order repaired and checked by reading, not compiled, not run on hardware
//
#include <hip/hip_runtime.h>
#include <hip/hip_bf16.h>

typedef _Float16 h8   __attribute__((ext_vector_type(8)));
typedef _Float16 v16h __attribute__((ext_vector_type(16)));
typedef float    v8f  __attribute__((ext_vector_type(8)));

#define BATCH    65536
#define OBS_D    55

// d_ws layout (units: _Float16). All panels transposed to [N][K]: rows are the
// WMMA A-matrix rows (output features), K contiguous.
#define W1T_OFF  0        // [256][64]  (K padded 58->64 with zeros)
#define W2T_OFF  16384    // [256][256]
#define RWT_OFF  81920    // [256][256]
#define HWT_OFF  147456   // [16][256]  rows 0-3 = mean_w, 4-7 = logstd_w, 8-15 = 0
#define WS_HALF_TOTAL 151552

// dynamic LDS layout (bytes)
#define LDS_W1T_BYTES  32768               // phi_w1^T, resident whole kernel
#define LDS_WBUF_OFF   32768
#define LDS_WBUF_BYTES 131072              // w2^T during perms; rho_w1^T afterwards
#define LDS_PAIR_OFF   (LDS_WBUF_OFF + LDS_WBUF_BYTES)   // 163840
#define RAW_LD   84                        // f32 per staged row (padded)
#define INP_LD   72                        // f16 per perm-input row (58 used, padded)
#define ACT_LD   264                       // f16 per activation row (256 used, padded)
#define RAW_BYTES (16*RAW_LD*4)            // 5376
#define INP_BYTES (16*INP_LD*2)            // 2304
#define ACT_BYTES (16*ACT_LD*2)            // 8448
#define PAIR_BYTES (RAW_BYTES+INP_BYTES+ACT_BYTES)       // 16128
#define BIAS_OFF  (LDS_PAIR_OFF + 4*PAIR_BYTES)          // 228352 (32B aligned)
#define SMEM_BYTES (BIAS_OFF + 3*1024)                   // 231424 (<= 320KB WGP LDS)

// ---------------------------------------------------------------------------
// Weight preprocessing: f32 -> f16 transposed panels in d_ws
// ---------------------------------------------------------------------------
__global__ void prep_weights(const float* __restrict__ w1, const float* __restrict__ w2,
                             const float* __restrict__ rw, const float* __restrict__ mw,
                             const float* __restrict__ lw, _Float16* __restrict__ ws) {
  int idx = blockIdx.x * blockDim.x + threadIdx.x;
  if (idx >= WS_HALF_TOTAL) return;
  float v;
  if (idx < W2T_OFF) {                       // w1^T [256][64], K pad
    int n = idx >> 6, k = idx & 63;
    v = (k < 58) ? w1[k * 256 + n] : 0.f;
  } else if (idx < RWT_OFF) {                // w2^T [256][256]
    int t = idx - W2T_OFF; int n = t >> 8, k = t & 255;
    v = w2[k * 256 + n];
  } else if (idx < HWT_OFF) {                // rho_w1^T [256][256]
    int t = idx - RWT_OFF; int n = t >> 8, k = t & 255;
    v = rw[k * 256 + n];
  } else {                                   // packed heads [16][256]
    int t = idx - HWT_OFF; int n = t >> 8, k = t & 255;
    v = (n < 4) ? mw[k * 4 + n] : ((n < 8) ? lw[k * 4 + (n - 4)] : 0.f);
  }
  ws[idx] = (_Float16)v;
}

// ---------------------------------------------------------------------------
// WMMA helpers (gfx1250 v_wmma_f32_16x16x32_f16; M/N per ISA 7.12.2; A and B both use k = 16*half + s).
// We compute D = Oᵀ: A = weight tile [n][k], B = activation tile (k x m),
// so D's lane = batch column m, VGPRs walk 8 consecutive output features n.
// ---------------------------------------------------------------------------
__device__ __forceinline__ v8f wmma16(v16h a, v16h b, v8f c) {
  return __builtin_amdgcn_wmma_f32_16x16x32_f16(false, a, false, b, (short)0, c,
                                                false, false);
}

// A fragment (16x32 f16): lane holds row M = lane%16; K = kbase+half*16+0..15
// -> one contiguous 32B per-lane read from a row-major row.
__device__ __forceinline__ v16h afrag(const _Float16* row, int kbase, int half) {
  h8 lo = *(const h8*)(row + kbase + half * 16);
  h8 hi = *(const h8*)(row + kbase + half * 16 + 8);
  return __builtin_shufflevector(lo, hi, 0, 1, 2, 3, 4, 5, 6, 7,
                                 8, 9, 10, 11, 12, 13, 14, 15);
}

// B fragment (32x16 f16): lane holds column N = lane%16, K = half*16+0..15 ->
// one contiguous 32B per-lane load from a row-major [m][k] activation row.
__device__ __forceinline__ v16h bfrag(const _Float16* p) {
  h8 lo = *(const h8*)p;
  h8 hi = *(const h8*)(p + 8);
  return __builtin_shufflevector(lo, hi, 0, 1, 2, 3, 4, 5, 6, 7,
                                 8, 9, 10, 11, 12, 13, 14, 15);
}

#define ZERO8 (v8f){0.f,0.f,0.f,0.f,0.f,0.f,0.f,0.f}

// ---------------------------------------------------------------------------
// Fused actor kernel: a wave PAIR owns a 16-row batch tile; each wave computes
// half of the output features of every layer (pooled accum = 64 VGPRs/wave).
// ---------------------------------------------------------------------------
__global__ __launch_bounds__(256) void actor_kernel(
    const float* __restrict__ obs, const float* __restrict__ ag,
    const float* __restrict__ g,
    const float* __restrict__ phi_b1, const float* __restrict__ phi_b2,
    const float* __restrict__ rho_b1, const float* __restrict__ mean_b,
    const float* __restrict__ logstd_b,
    const _Float16* __restrict__ wsp, float* __restrict__ out) {
  extern __shared__ char smem[];
  _Float16* w1t  = (_Float16*)smem;
  _Float16* wbuf = (_Float16*)(smem + LDS_WBUF_OFF);
  float*    bls  = (float*)(smem + BIAS_OFF);   // [b1 | b2 | br], 256 f32 each
  const int tid   = threadIdx.x;
  const int wave  = tid >> 5;
  const int lane  = tid & 31;
  const int half  = lane >> 4;
  const int nn    = lane & 15;
  const int pair  = wave >> 1;
  const int nhalf = wave & 1;            // which half of the 256 features

  char* wbase = smem + LDS_PAIR_OFF + pair * PAIR_BYTES;
  float*    raw  = (float*)wbase;
  _Float16* inp  = (_Float16*)(wbase + RAW_BYTES);
  _Float16* actA = (_Float16*)(wbase + RAW_BYTES + INP_BYTES);

  // Block-cooperative: phi_w1^T (32KB) + w2^T (128KB) + biases into LDS.
  {
    const uint4* s1 = (const uint4*)(wsp + W1T_OFF);
    uint4* d1 = (uint4*)w1t;
    for (int t = tid; t < LDS_W1T_BYTES / 16; t += 256) d1[t] = s1[t];
    const uint4* s2 = (const uint4*)(wsp + W2T_OFF);
    uint4* d2 = (uint4*)wbuf;
    for (int t = tid; t < LDS_WBUF_BYTES / 16; t += 256) d2[t] = s2[t];
    for (int t = tid; t < 768; t += 256)
      bls[t] = (t < 256) ? phi_b1[t] : ((t < 512) ? phi_b2[t - 256] : rho_b1[t - 512]);
  }

  const int m0 = (blockIdx.x * 4 + pair) * 16;

  // Stage the pair's 16 raw rows (split across both waves):
  // obs[0..54] -> cols 0..54, ag -> 56..64, g -> 66..74
  for (int idx = nhalf * 32 + lane; idx < 16 * 73; idx += 64) {
    int r = idx / 73, c = idx % 73;
    int row = m0 + r;
    float v; int cc;
    if (c < 55)      { v = obs[row * OBS_D + c];   cc = c; }
    else if (c < 64) { v = ag[row * 9 + (c - 55)]; cc = 56 + (c - 55); }
    else             { v = g[row * 9 + (c - 64)];  cc = 66 + (c - 64); }
    raw[r * RAW_LD + cc] = v;
  }
  __syncthreads();

  v8f pooled[8];                          // this wave's 8 N-tiles
#pragma unroll
  for (int q = 0; q < 8; ++q) pooled[q] = ZERO8;

  for (int p = 0; p < 6; ++p) {
    const int oi = p >> 1;
    const int oj = (0x102021 >> (4 * p)) & 0xF;   // {1,2,0,2,0,1}

    // Build 64-wide (58 + zero pad) permuted input rows in f16 (pair-split).
    for (int idx = nhalf * 32 + lane; idx < 16 * 64; idx += 64) {
      int r = idx >> 6, c = idx & 63;
      const float* rw = raw + r * RAW_LD;
      float v = 0.f;
      if (c < 3)        v = rw[56 + 3 * oi + c];
      else if (c < 6)   v = rw[56 + 3 * oj + (c - 3)];
      else if (c < 9)   v = rw[66 + 3 * oi + (c - 6)];
      else if (c < 12)  v = rw[66 + 3 * oj + (c - 9)];
      else if (c < 22)  v = rw[c - 12];                       // obs_body
      else if (c < 40)  { int d = c - 22;
                          v = (d < 3) ? ((d == oi) ? 1.f : 0.f)
                                      : rw[10 + 15 * oi + (d - 3)]; }
      else if (c < 58)  { int d = c - 40;
                          v = (d < 3) ? ((d == oj) ? 1.f : 0.f)
                                      : rw[10 + 15 * oj + (d - 3)]; }
      inp[r * INP_LD + c] = (_Float16)v;
    }
    __syncthreads();

    // phi layer 1: h^T tiles = w1-tile x inp-frag; wave owns nt = nhalf*8+q.
    // Epilogue: 8 consecutive features per lane -> ONE contiguous b128 store.
    {
      v16h b0 = bfrag(inp + nn * INP_LD + half * 16);
      v16h b1 = bfrag(inp + nn * INP_LD + 32 + half * 16);
      for (int q = 0; q < 8; ++q) {
        int nt = nhalf * 8 + q;
        const _Float16* wrow = w1t + (nt * 16 + nn) * 64;
        v8f c = ZERO8;
        c = wmma16(afrag(wrow, 0, half), b0, c);
        c = wmma16(afrag(wrow, 32, half), b1, c);
        v8f bb = *(const v8f*)(bls + nt * 16 + 8 * half);
        h8 ph;
#pragma unroll
        for (int v = 0; v < 8; ++v) {
          float t = c[v] + bb[v]; t = t > 0.f ? t : 0.f;
          ph[v] = (_Float16)t;
        }
        *(h8*)(actA + nn * ACT_LD + nt * 16 + 8 * half) = ph;
      }
    }
    __syncthreads();

    // phi layer 2: pooled += relu(w2-tile x h-frag + b2).  B (act) fragment is
    // shared across the 4-tile N-chunk; all operands from LDS.
    {
      const _Float16* arow = actA + nn * ACT_LD;
#pragma unroll
      for (int ntc = 0; ntc < 2; ++ntc) {
        v8f acc[4] = {ZERO8, ZERO8, ZERO8, ZERO8};
        for (int kc = 0; kc < 8; ++kc) {
          v16h b = bfrag(arow + kc * 32 + half * 16);
#pragma unroll
          for (int q2 = 0; q2 < 4; ++q2) {
            int nt = nhalf * 8 + ntc * 4 + q2;
            acc[q2] = wmma16(afrag(wbuf + (nt * 16 + nn) * 256, kc * 32, half),
                             b, acc[q2]);
          }
        }
#pragma unroll
        for (int q2 = 0; q2 < 4; ++q2) {
          int nt = nhalf * 8 + ntc * 4 + q2;
          v8f bb = *(const v8f*)(bls + 256 + nt * 16 + 8 * half);
#pragma unroll
          for (int v = 0; v < 8; ++v) {
            float t = acc[q2][v] + bb[v]; t = t > 0.f ? t : 0.f;
            pooled[ntc * 4 + q2][v] += t;
          }
        }
      }
    }
    __syncthreads();
  }

  // pooled (f32 regs) -> f16 panel in actA (contiguous b128 stores), then
  // recycle the pooled register file as the rho accumulators.
#pragma unroll
  for (int q = 0; q < 8; ++q) {
    int nt = nhalf * 8 + q;
    h8 ph;
#pragma unroll
    for (int v = 0; v < 8; ++v) ph[v] = (_Float16)pooled[q][v];
    *(h8*)(actA + nn * ACT_LD + nt * 16 + 8 * half) = ph;
    pooled[q] = ZERO8;
  }
  __syncthreads();   // all waves done reading w2^T panel

  // Block-cooperative: overwrite wbuf with rho_w1^T (128KB).
  {
    const uint4* s2 = (const uint4*)(wsp + RWT_OFF);
    uint4* d2 = (uint4*)wbuf;
    for (int t = tid; t < LDS_WBUF_BYTES / 16; t += 256) d2[t] = s2[t];
  }
  __syncthreads();

  // rho: K-outer, all 8 of this wave's accumulators live; act B-fragment
  // loaded once per kc and shared across the 8 weight tiles.
  {
    const _Float16* arow = actA + nn * ACT_LD;
    for (int kc = 0; kc < 8; ++kc) {
      v16h b = bfrag(arow + kc * 32 + half * 16);
#pragma unroll
      for (int q = 0; q < 8; ++q) {
        int nt = nhalf * 8 + q;
        pooled[q] = wmma16(afrag(wbuf + (nt * 16 + nn) * 256, kc * 32, half),
                           b, pooled[q]);
      }
    }
  }
  __syncthreads();   // partner must finish reading actA before x overwrites it
#pragma unroll
  for (int q = 0; q < 8; ++q) {
    int nt = nhalf * 8 + q;
    v8f bb = *(const v8f*)(bls + 512 + nt * 16 + 8 * half);
    h8 ph;
#pragma unroll
    for (int v = 0; v < 8; ++v) {
      float t = pooled[q][v] + bb[v]; t = t > 0.f ? t : 0.f;
      ph[v] = (_Float16)t;
    }
    *(h8*)(actA + nn * ACT_LD + nt * 16 + 8 * half) = ph;
  }
  __syncthreads();

  // Heads (one wave per pair): D tile = headsᵀ; half=0 lanes hold features
  // n=0..7 (4 mean + 4 logstd) for batch row m0+nn -> two float4 stores.
  if (nhalf == 0) {
    const _Float16* hwt = wsp + HWT_OFF;
    const _Float16* arow = actA + nn * ACT_LD;
    v8f c = ZERO8;
    for (int kc = 0; kc < 8; ++kc) {
      v16h b = bfrag(arow + kc * 32 + half * 16);
      c = wmma16(afrag(hwt + nn * 256, kc * 32, half), b, c);
    }
    if (half == 0) {
      float4 mb = *(const float4*)mean_b;
      float4 lb = *(const float4*)logstd_b;
      int row = m0 + nn;
      float4 mo, lo;
      mo.x = c[0] + mb.x; mo.y = c[1] + mb.y;
      mo.z = c[2] + mb.z; mo.w = c[3] + mb.w;
      lo.x = fminf(fmaxf(c[4] + lb.x, -20.f), 2.f);
      lo.y = fminf(fmaxf(c[5] + lb.y, -20.f), 2.f);
      lo.z = fminf(fmaxf(c[6] + lb.z, -20.f), 2.f);
      lo.w = fminf(fmaxf(c[7] + lb.w, -20.f), 2.f);
      *(float4*)(out + row * 4) = mo;
      *(float4*)(out + BATCH * 4 + row * 4) = lo;
    }
  }
}

// ---------------------------------------------------------------------------
extern "C" void kernel_launch(void* const* d_in, const int* in_sizes, int n_in,
                              void* d_out, int out_size, void* d_ws, size_t ws_size,
                              hipStream_t stream) {
  const float* obs      = (const float*)d_in[0];
  const float* ag       = (const float*)d_in[1];
  const float* g        = (const float*)d_in[2];
  const float* phi_w1   = (const float*)d_in[3];
  const float* phi_b1   = (const float*)d_in[4];
  const float* phi_w2   = (const float*)d_in[5];
  const float* phi_b2   = (const float*)d_in[6];
  const float* rho_w1   = (const float*)d_in[7];
  const float* rho_b1   = (const float*)d_in[8];
  const float* mean_w   = (const float*)d_in[9];
  const float* mean_b   = (const float*)d_in[10];
  const float* logstd_w = (const float*)d_in[11];
  const float* logstd_b = (const float*)d_in[12];
  _Float16* ws = (_Float16*)d_ws;
  float* out = (float*)d_out;

  prep_weights<<<(WS_HALF_TOTAL + 255) / 256, 256, 0, stream>>>(
      phi_w1, phi_w2, rho_w1, mean_w, logstd_w, ws);

  (void)hipFuncSetAttribute((const void*)actor_kernel,
                            hipFuncAttributeMaxDynamicSharedMemorySize,
                            SMEM_BYTES);
  actor_kernel<<<BATCH / 64, 256, SMEM_BYTES, stream>>>(
      obs, ag, g, phi_b1, phi_b2, rho_b1, mean_b, logstd_b, ws, out);
}
